// MambaBlock_71751723647176
// MI455X (gfx1250) — matrix operand order repaired and checked by reading, not compiled, not run on hardware
//
#include <hip/hip_runtime.h>
#include <hip/hip_bf16.h>
#include <cstdint>

#define D_MODEL 256
#define D_INNER 512
#define D_STATE 32
#define DT_RANK 16
#define IN_DIM  64
#define BATCH   4
#define HWDIM   48
#define SEQLEN  (HWDIM*HWDIM)     // 2304
#define MTOT    (BATCH*SEQLEN)    // 9216
#define KCONV   (IN_DIM*9)        // 576
#define XDBL_N  (DT_RANK + 2*D_STATE)  // 80
#define CHUNK   64
#define NCHUNK  (SEQLEN / CHUNK)  // 36

typedef __attribute__((ext_vector_type(16))) __bf16   v16bf;
typedef __attribute__((ext_vector_type(8)))  float    v8f;
typedef __attribute__((ext_vector_type(8)))  unsigned v8u;

// ---------- helpers ----------
__device__ __forceinline__ unsigned short f2bf(float x) {
  unsigned u = __builtin_bit_cast(unsigned, x);
  u += 0x7FFFu + ((u >> 16) & 1u);           // round-to-nearest-even
  return (unsigned short)(u >> 16);
}
__device__ __forceinline__ float sigmoidf_(float x) { return 1.0f / (1.0f + __expf(-x)); }

// A fragment 16x32 bf16: lane&15 = M row, lane>>4 selects K half.
__device__ __forceinline__ v16bf load_a_frag(const unsigned short* __restrict__ A,
                                             int K, int m, int k0, int lane) {
  const unsigned* rp = (const unsigned*)(A + (size_t)m * K + k0);
  int o = 4 * (lane >> 4);
  v8u a;
#pragma unroll
  for (int v = 0; v < 4; ++v) a[v] = rp[o + v];
#pragma unroll
  for (int v = 0; v < 4; ++v) a[4 + v] = rp[8 + o + v];
  return __builtin_bit_cast(v16bf, a);
}

// B fragment 32x16 bf16 from K-major weights Bt[K][N]: column n0, elems = K rows k..k+7, k+16..k+23.
__device__ __forceinline__ v16bf load_b_frag(const unsigned short* __restrict__ Bt,
                                             int N, int k, int n0) {
  v16bf b;
#pragma unroll
  for (int s = 0; s < 16; ++s) {
    int kk = k + (s < 8 ? s : 16 + (s - 8));
    b[s] = __builtin_bit_cast(__bf16, Bt[(size_t)kk * N + n0]);
  }
  return b;
}

// ============ blocked WMMA GEMM: block tile 256(M) x 64(N), wave tile 32x64 ============
// B 32x64 K-step tile staged to LDS with async copies, double-buffered.
// EPI 0: f32 store; EPI 1: BN+ReLU -> bf16; EPI 2: transposed store [b, n, l]
template <int EPI>
__global__ __launch_bounds__(256) void wmma_gemm_blocked(
    const unsigned short* __restrict__ A, const unsigned short* __restrict__ Bt,
    float* __restrict__ outF, unsigned short* __restrict__ outBF,
    const float* __restrict__ e0, const float* __restrict__ e1,
    const float* __restrict__ e2, const float* __restrict__ e3,
    int M, int N, int K, int nBlkN) {
  __shared__ unsigned short tile[2][32 * 64];   // 2 x 4KB

  const int t    = threadIdx.x;
  const int w    = t >> 5;
  const int lane = t & 31;
  const int bn   = blockIdx.x % nBlkN;
  const int bm   = blockIdx.x / nBlkN;
  const int n_blk = bn * 64;
  const int m_blk = bm * 256;

  // staging: 256 threads x 16B = 4KB tile (rows = K within step, cols = 64 bf16)
  const int srow = t >> 3;           // 0..31
  const int scol = (t & 7) * 8;      // 8 bf16 = 16B per thread
  unsigned ldsOff0 = (unsigned)(uintptr_t)&tile[0][srow * 64 + scol];
  unsigned ldsOff1 = (unsigned)(uintptr_t)&tile[1][srow * 64 + scol];
  const int nk = K >> 5;

  // stage K-step 0 into buffer 0
  {
    unsigned long long ga =
        (unsigned long long)(uintptr_t)(Bt + (size_t)srow * N + n_blk + scol);
    asm volatile("global_load_async_to_lds_b128 %0, %1, off"
                 :: "v"(ldsOff0), "v"(ga) : "memory");
  }

  v8f acc[2][4] = {};
  const int mA0 = m_blk + w * 32 + (lane & 15);

  for (int ks = 0; ks < nk; ++ks) {
    const int cur = ks & 1;
    if (ks + 1 < nk) {   // prefetch next K-step into the other buffer
      unsigned long long ga = (unsigned long long)(uintptr_t)
          (Bt + (size_t)((ks + 1) * 32 + srow) * N + n_blk + scol);
      unsigned dst = (cur == 0) ? ldsOff1 : ldsOff0;
      asm volatile("global_load_async_to_lds_b128 %0, %1, off"
                   :: "v"(dst), "v"(ga) : "memory");
      asm volatile("s_wait_asynccnt 1" ::: "memory");  // current tile done, next in flight
    } else {
      asm volatile("s_wait_asynccnt 0" ::: "memory");
    }
    __syncthreads();     // whole tile[cur] visible to all waves

    v16bf fa0 = load_a_frag(A, K, mA0,      ks * 32, lane);
    v16bf fa1 = load_a_frag(A, K, mA0 + 16, ks * 32, lane);
#pragma unroll
    for (int j = 0; j < 4; ++j) {
      v16bf fb;
#pragma unroll
      for (int s = 0; s < 16; ++s) {
        int kk = 8 * (lane >> 4) + (s < 8 ? s : 16 + (s - 8));
        fb[s] = __builtin_bit_cast(__bf16, tile[cur][kk * 64 + j * 16 + (lane & 15)]);
      }
      acc[0][j] = __builtin_amdgcn_wmma_f32_16x16x32_bf16(false, fa0, false, fb,
                                                          (short)0, acc[0][j], false, false);
      acc[1][j] = __builtin_amdgcn_wmma_f32_16x16x32_bf16(false, fa1, false, fb,
                                                          (short)0, acc[1][j], false, false);
    }
    __syncthreads();     // reads of tile[cur] done before it is restaged
  }

#pragma unroll
  for (int mi = 0; mi < 2; ++mi) {
    int mBase = m_blk + w * 32 + mi * 16 + 8 * (lane >> 4);
#pragma unroll
    for (int j = 0; j < 4; ++j) {
      int n = n_blk + j * 16 + (lane & 15);
      if (EPI == 0) {
#pragma unroll
        for (int r = 0; r < 8; ++r) outF[(size_t)(mBase + r) * N + n] = acc[mi][j][r];
      } else if (EPI == 1) {
        float inv  = e0[n] * rsqrtf(e3[n] + 1e-5f);
        float bias = e1[n] - e2[n] * inv;
#pragma unroll
        for (int r = 0; r < 8; ++r) {
          float v = acc[mi][j][r] * inv + bias;
          v = v > 0.f ? v : 0.f;
          outBF[(size_t)(mBase + r) * N + n] = f2bf(v);
        }
      } else {
#pragma unroll
        for (int r = 0; r < 8; ++r) {
          int m = mBase + r;
          int b = m / SEQLEN, l = m % SEQLEN;
          outF[((size_t)b * N + n) * SEQLEN + l] = acc[mi][j][r];
        }
      }
    }
  }
}

// ============ simple per-wave GEMM (for N=80 x_proj), f32 out ============
__global__ void wmma_gemm_simple(const unsigned short* __restrict__ A,
                                 const unsigned short* __restrict__ Bt,
                                 float* __restrict__ outF,
                                 int M, int N, int K, int tilesN) {
  int wid  = blockIdx.x * (blockDim.x >> 5) + (threadIdx.x >> 5);
  int lane = threadIdx.x & 31;
  int tm = wid / tilesN, tn = wid % tilesN;
  int m0 = tm * 16, n0 = tn * 16;
  if (m0 >= M) return;
  int mA = m0 + (lane & 15);
  v8f acc = {};
  for (int k0 = 0; k0 < K; k0 += 32) {
    v16bf fa = load_a_frag(A, K, mA, k0, lane);
    v16bf fb = load_b_frag(Bt, N, k0 + 8 * (lane >> 4), n0 + (lane & 15));
    acc = __builtin_amdgcn_wmma_f32_16x16x32_bf16(false, fa, false, fb,
                                                  (short)0, acc, false, false);
  }
  int n = n0 + (lane & 15);
  int mBase = m0 + 8 * (lane >> 4);
#pragma unroll
  for (int r = 0; r < 8; ++r) outF[(size_t)(mBase + r) * N + n] = acc[r];
}

// ---------- im2col (conv3x3 pad1) -> bf16 [MTOT, KCONV] ----------
__global__ void im2col_kernel(const float* __restrict__ x, unsigned short* __restrict__ out) {
  int idx = blockIdx.x * blockDim.x + threadIdx.x;
  if (idx >= MTOT * KCONV) return;
  int m = idx / KCONV, k = idx % KCONV;
  int ci = k / 9, kk = k % 9, ky = kk / 3, kx = kk % 3;
  int b = m / SEQLEN, l = m % SEQLEN, yy = l / HWDIM, xx = l % HWDIM;
  int iy = yy + ky - 1, ix = xx + kx - 1;
  float v = 0.f;
  if ((unsigned)iy < (unsigned)HWDIM && (unsigned)ix < (unsigned)HWDIM)
    v = x[(((size_t)b * IN_DIM + ci) * HWDIM + iy) * HWDIM + ix];
  out[idx] = f2bf(v);
}

// ---------- weight convert: W[N,K] f32 -> Wt[K,N] bf16 ----------
__global__ void wcvt_t_kernel(const float* __restrict__ W, unsigned short* __restrict__ out,
                              int N, int K) {
  int idx = blockIdx.x * blockDim.x + threadIdx.x;
  if (idx >= N * K) return;
  int n = idx / K, k = idx % K;
  out[(size_t)k * N + n] = f2bf(W[idx]);
}

// ---------- depthwise causal conv1d (k=4) + SiLU over u = xz[:, :512] ----------
__global__ void dwconv_silu_kernel(const float* __restrict__ xz, const float* __restrict__ w,
                                   const float* __restrict__ bias,
                                   float* __restrict__ uF, unsigned short* __restrict__ uBF) {
  int idx = blockIdx.x * blockDim.x + threadIdx.x;
  if (idx >= MTOT * D_INNER) return;
  int m = idx / D_INNER, d = idx % D_INNER;
  int l = m % SEQLEN;
  float acc = bias[d];
#pragma unroll
  for (int j = 0; j < 4; ++j) {
    int ll = l - 3 + j;
    if (ll >= 0) acc += w[d * 4 + j] * xz[(size_t)(m - 3 + j) * (2 * D_INNER) + d];
  }
  float s = acc * sigmoidf_(acc);
  uF[idx]  = s;
  uBF[idx] = f2bf(s);
}

// ---------- dt_proj (K=16) + softplus ----------
__global__ void dt_softplus_kernel(const float* __restrict__ xdbl, const float* __restrict__ dtw,
                                   const float* __restrict__ dtb, float* __restrict__ delta) {
  int idx = blockIdx.x * blockDim.x + threadIdx.x;
  if (idx >= MTOT * D_INNER) return;
  int m = idx / D_INNER, d = idx % D_INNER;
  const float* xr = xdbl + (size_t)m * XDBL_N;
  float acc = dtb[d];
#pragma unroll
  for (int r = 0; r < DT_RANK; ++r) acc += xr[r] * dtw[d * DT_RANK + r];
  delta[idx] = (acc > 20.f) ? acc : log1pf(__expf(acc));
}

// ============ chunked selective scan (3 passes), lane = state index ============
// pass1: per (b,d,chunk): local state from s=0 plus sum(dt) over chunk
__global__ void scan_pass1(const float* __restrict__ delta, const float* __restrict__ xdbl,
                           const float* __restrict__ uF, const float* __restrict__ A_log,
                           float* __restrict__ Sloc, float* __restrict__ sumdt) {
  int wid  = blockIdx.x * (blockDim.x >> 5) + (threadIdx.x >> 5);
  int lane = threadIdx.x & 31;
  if (wid >= BATCH * D_INNER * NCHUNK) return;
  int c = wid % NCHUNK, bd = wid / NCHUNK;
  int b = bd / D_INNER, d = bd % D_INNER;
  float Ad = -__expf(A_log[d * D_STATE + lane]);
  size_t row0 = (size_t)b * SEQLEN + (size_t)c * CHUNK;
  const float* dl = delta + row0 * D_INNER + d;
  const float* ul = uF    + row0 * D_INNER + d;
  const float* xd = xdbl  + row0 * XDBL_N;
  float s = 0.f, sd = 0.f;
  for (int l = 0; l < CHUNK; ++l) {
    float dt = dl[(size_t)l * D_INNER];
    float uu = ul[(size_t)l * D_INNER];
    float Bv = xd[(size_t)l * XDBL_N + DT_RANK + lane];
    s = s * __expf(dt * Ad) + dt * uu * Bv;
    sd += dt;
  }
  Sloc[((size_t)bd * NCHUNK + c) * D_STATE + lane] = s;
  if (lane == 0) sumdt[(size_t)bd * NCHUNK + c] = sd;
}

// pass2: per (b,d): combine chunks sequentially (chunk decay = exp(A * sum(dt)))
__global__ void scan_pass2(const float* __restrict__ Sloc, const float* __restrict__ sumdt,
                           const float* __restrict__ A_log, float* __restrict__ sstart) {
  int wid  = blockIdx.x * (blockDim.x >> 5) + (threadIdx.x >> 5);
  int lane = threadIdx.x & 31;
  if (wid >= BATCH * D_INNER) return;
  int d = wid % D_INNER;
  float Ad = -__expf(A_log[d * D_STATE + lane]);
  float s = 0.f;
  for (int c = 0; c < NCHUNK; ++c) {
    size_t ix = ((size_t)wid * NCHUNK + c);
    sstart[ix * D_STATE + lane] = s;
    s = s * __expf(Ad * sumdt[ix]) + Sloc[ix * D_STATE + lane];
  }
}

// pass3: per (b,d,chunk): replay chunk from known start state, emit y
__global__ void scan_pass3(const float* __restrict__ delta, const float* __restrict__ xdbl,
                           const float* __restrict__ uF, const float* __restrict__ A_log,
                           const float* __restrict__ Dp, const float* __restrict__ sstart,
                           float* __restrict__ ypre) {
  int wid  = blockIdx.x * (blockDim.x >> 5) + (threadIdx.x >> 5);
  int lane = threadIdx.x & 31;
  if (wid >= BATCH * D_INNER * NCHUNK) return;
  int c = wid % NCHUNK, bd = wid / NCHUNK;
  int b = bd / D_INNER, d = bd % D_INNER;
  float Ad = -__expf(A_log[d * D_STATE + lane]);
  float Dd = Dp[d];
  float s  = sstart[((size_t)bd * NCHUNK + c) * D_STATE + lane];
  size_t row0 = (size_t)b * SEQLEN + (size_t)c * CHUNK;
  const float* dl = delta + row0 * D_INNER + d;
  const float* ul = uF    + row0 * D_INNER + d;
  const float* xd = xdbl  + row0 * XDBL_N;
  float* yl       = ypre  + row0 * D_INNER + d;
  for (int l = 0; l < CHUNK; ++l) {
    float dt = dl[(size_t)l * D_INNER];
    float uu = ul[(size_t)l * D_INNER];
    float Bv = xd[(size_t)l * XDBL_N + DT_RANK + lane];
    float Cv = xd[(size_t)l * XDBL_N + DT_RANK + D_STATE + lane];
    s = s * __expf(dt * Ad) + dt * uu * Bv;
    float cc = s * Cv;
#pragma unroll
    for (int off = 16; off > 0; off >>= 1) cc += __shfl_xor(cc, off, 32);
    if (lane == 0) yl[(size_t)l * D_INNER] = cc + uu * Dd;
  }
}

// ---------- gating: ybf = bf16( ypre * silu(z) ) ----------
__global__ void gate_kernel(const float* __restrict__ ypre, const float* __restrict__ xz,
                            unsigned short* __restrict__ ybf) {
  int idx = blockIdx.x * blockDim.x + threadIdx.x;
  if (idx >= MTOT * D_INNER) return;
  int m = idx / D_INNER, d = idx % D_INNER;
  float z = xz[(size_t)m * (2 * D_INNER) + D_INNER + d];
  float y = ypre[idx] * (z * sigmoidf_(z));
  ybf[idx] = f2bf(y);
}

// ---------- launcher ----------
extern "C" void kernel_launch(void* const* d_in, const int* in_sizes, int n_in,
                              void* d_out, int out_size, void* d_ws, size_t ws_size,
                              hipStream_t stream) {
  const float* x         = (const float*)d_in[0];
  const float* conv_w    = (const float*)d_in[1];
  const float* bn_gamma  = (const float*)d_in[2];
  const float* bn_beta   = (const float*)d_in[3];
  const float* bn_mean   = (const float*)d_in[4];
  const float* bn_var    = (const float*)d_in[5];
  const float* in_proj_w = (const float*)d_in[6];
  const float* conv1d_w  = (const float*)d_in[7];
  const float* conv1d_b  = (const float*)d_in[8];
  const float* x_proj_w  = (const float*)d_in[9];
  const float* dt_proj_w = (const float*)d_in[10];
  const float* dt_proj_b = (const float*)d_in[11];
  const float* A_log     = (const float*)d_in[12];
  const float* Dp        = (const float*)d_in[13];
  const float* out_proj_w= (const float*)d_in[14];
  float* out = (float*)d_out;

  char* base = (char*)d_ws;
  size_t off = 0;
  auto alloc = [&](size_t bytes) -> void* {
    void* p = base + off;
    off = (off + bytes + 255) & ~(size_t)255;
    return p;
  };
  unsigned short* im2col = (unsigned short*)alloc((size_t)MTOT * KCONV * 2);
  unsigned short* wconv_t= (unsigned short*)alloc((size_t)KCONV * D_MODEL * 2);
  unsigned short* seq_bf = (unsigned short*)alloc((size_t)MTOT * D_MODEL * 2);
  unsigned short* win_t  = (unsigned short*)alloc((size_t)D_MODEL * 2 * D_INNER * 2);
  float*          xz     = (float*)alloc((size_t)MTOT * 2 * D_INNER * 4);
  float*          uF     = (float*)alloc((size_t)MTOT * D_INNER * 4);
  unsigned short* uBF    = (unsigned short*)alloc((size_t)MTOT * D_INNER * 2);
  unsigned short* wx_t   = (unsigned short*)alloc((size_t)D_INNER * XDBL_N * 2);
  float*          xdbl   = (float*)alloc((size_t)MTOT * XDBL_N * 4);
  float*          delta  = (float*)alloc((size_t)MTOT * D_INNER * 4);
  float*          ypre   = (float*)alloc((size_t)MTOT * D_INNER * 4);
  unsigned short* ybf    = (unsigned short*)alloc((size_t)MTOT * D_INNER * 2);
  unsigned short* wout_t = (unsigned short*)alloc((size_t)D_INNER * D_MODEL * 2);
  float*          Sloc   = (float*)alloc((size_t)BATCH * D_INNER * NCHUNK * D_STATE * 4);
  float*          sstart = (float*)alloc((size_t)BATCH * D_INNER * NCHUNK * D_STATE * 4);
  float*          sumdt  = (float*)alloc((size_t)BATCH * D_INNER * NCHUNK * 4);
  (void)ws_size; (void)in_sizes; (void)n_in; (void)out_size;

  const int T = 256;
  auto cdiv = [](long long a, long long b) { return (int)((a + b - 1) / b); };

  // weight preparation (transpose + f32->bf16)
  wcvt_t_kernel<<<cdiv((long long)D_MODEL * KCONV, T), T, 0, stream>>>(conv_w, wconv_t, D_MODEL, KCONV);
  wcvt_t_kernel<<<cdiv((long long)2 * D_INNER * D_MODEL, T), T, 0, stream>>>(in_proj_w, win_t, 2 * D_INNER, D_MODEL);
  wcvt_t_kernel<<<cdiv((long long)XDBL_N * D_INNER, T), T, 0, stream>>>(x_proj_w, wx_t, XDBL_N, D_INNER);
  wcvt_t_kernel<<<cdiv((long long)D_MODEL * D_INNER, T), T, 0, stream>>>(out_proj_w, wout_t, D_MODEL, D_INNER);

  // conv3x3 as im2col + blocked WMMA GEMM with BN+ReLU epilogue -> seq_bf [MTOT, 256]
  im2col_kernel<<<cdiv((long long)MTOT * KCONV, T), T, 0, stream>>>(x, im2col);
  {
    int nBlkN = D_MODEL / 64;
    wmma_gemm_blocked<1><<<(MTOT / 256) * nBlkN, 256, 0, stream>>>(
        im2col, wconv_t, nullptr, seq_bf, bn_gamma, bn_beta, bn_mean, bn_var,
        MTOT, D_MODEL, KCONV, nBlkN);
  }
  // in_proj: [MTOT,256] x [256,1024] -> xz f32
  {
    int nBlkN = (2 * D_INNER) / 64;
    wmma_gemm_blocked<0><<<(MTOT / 256) * nBlkN, 256, 0, stream>>>(
        seq_bf, win_t, xz, nullptr, nullptr, nullptr, nullptr, nullptr,
        MTOT, 2 * D_INNER, D_MODEL, nBlkN);
  }
  // depthwise causal conv1d + SiLU
  dwconv_silu_kernel<<<cdiv((long long)MTOT * D_INNER, T), T, 0, stream>>>(
      xz, conv1d_w, conv1d_b, uF, uBF);
  // x_proj: [MTOT,512] x [512,80] -> xdbl f32 (N=80 -> simple per-wave kernel)
  {
    int tilesN = XDBL_N / 16, waves = (MTOT / 16) * tilesN;
    wmma_gemm_simple<<<cdiv(waves, 8), 256, 0, stream>>>(
        uBF, wx_t, xdbl, MTOT, XDBL_N, D_INNER, tilesN);
  }
  // dt_proj + softplus
  dt_softplus_kernel<<<cdiv((long long)MTOT * D_INNER, T), T, 0, stream>>>(
      xdbl, dt_proj_w, dt_proj_b, delta);
  // chunked selective scan: 73728-wave passes around a tiny 36-step sequential pass
  scan_pass1<<<cdiv((long long)BATCH * D_INNER * NCHUNK * 32, 256), 256, 0, stream>>>(
      delta, xdbl, uF, A_log, Sloc, sumdt);
  scan_pass2<<<cdiv((long long)BATCH * D_INNER * 32, 256), 256, 0, stream>>>(
      Sloc, sumdt, A_log, sstart);
  scan_pass3<<<cdiv((long long)BATCH * D_INNER * NCHUNK * 32, 256), 256, 0, stream>>>(
      delta, xdbl, uF, A_log, Dp, sstart, ypre);
  // gating
  gate_kernel<<<cdiv((long long)MTOT * D_INNER, T), T, 0, stream>>>(ypre, xz, ybf);
  // out_proj with transposed store into [B, 256, H, W]
  {
    int nBlkN = D_MODEL / 64;
    wmma_gemm_blocked<2><<<(MTOT / 256) * nBlkN, 256, 0, stream>>>(
        ybf, wout_t, out, nullptr, nullptr, nullptr, nullptr, nullptr,
        MTOT, D_MODEL, D_INNER, nBlkN);
  }
}
